// PointNet2_25005299598071
// MI455X (gfx1250) — matrix operand order repaired and checked by reading, compiled, not run on hardware
//
#include <hip/hip_runtime.h>
#include <hip/hip_bf16.h>
#include <stdint.h>

// ---------------------------------------------------------------------------
// PointNet++ semantic-seg forward for MI455X (gfx1250, wave32, WMMA).
// All MLP stages run as f16-in / f32-accum WMMA GEMMs (v_wmma_f32_16x16x32_f16)
// with BN folded into a per-channel scale/shift epilogue + ReLU.
// GEMM is register-blocked: each wave produces a 32x64 output tile
// (8 accumulators, 8 WMMAs per K=32 step, ~22 FLOP/byte from L2).
// FPS / ball-query / 3-NN are scalar latency-bound kernels (tiny vs GEMM).
// ---------------------------------------------------------------------------

typedef _Float16 h16;
typedef _Float16 v8h  __attribute__((ext_vector_type(8)));
typedef _Float16 v16h __attribute__((ext_vector_type(16)));
typedef float    v8f  __attribute__((ext_vector_type(8)));

// ------------------------------ FPS ----------------------------------------
// One block per batch. dist[] lives in LDS (N<=8192 -> 32KB of 320KB WGP LDS).
__global__ void __launch_bounds__(256) k_fps(const float* __restrict__ xyz,
                                             int N, int npoint,
                                             int* __restrict__ out)
{
    __shared__ float dist[8192];
    __shared__ float rv[256];
    __shared__ int   ri[256];
    __shared__ float fpx[3];

    const int b  = blockIdx.x;
    const int t  = threadIdx.x;
    const int nt = blockDim.x;
    const float* p = xyz + (size_t)b * N * 3;

    for (int i = t; i < N; i += nt) dist[i] = 1e10f;
    __syncthreads();

    int far = 0;
    for (int it = 0; it < npoint; ++it) {
        if (t == 0) {
            out[b * npoint + it] = far;          // reference emits pre-update idx
            fpx[0] = p[far * 3 + 0];
            fpx[1] = p[far * 3 + 1];
            fpx[2] = p[far * 3 + 2];
        }
        __syncthreads();
        const float fx = fpx[0], fy = fpx[1], fz = fpx[2];
        float bv = -1.0f; int bi = 0x7fffffff;
        for (int i = t; i < N; i += nt) {
            float dx = p[i * 3 + 0] - fx;
            float dy = p[i * 3 + 1] - fy;
            float dz = p[i * 3 + 2] - fz;
            float d  = dx * dx + dy * dy + dz * dz;
            float m  = fminf(dist[i], d);
            dist[i]  = m;
            if (m > bv) { bv = m; bi = i; }      // first-max within a thread
        }
        rv[t] = bv; ri[t] = bi;
        __syncthreads();
        for (int s = nt >> 1; s > 0; s >>= 1) {
            if (t < s) {
                if (rv[t + s] > rv[t] ||
                    (rv[t + s] == rv[t] && ri[t + s] < ri[t])) {
                    rv[t] = rv[t + s]; ri[t] = ri[t + s];
                }
            }
            __syncthreads();
        }
        far = ri[0];
        __syncthreads();
    }
}

// ------------------------- gather sampled xyz ------------------------------
__global__ void k_gather_xyz(const float* __restrict__ xyz,
                             const int* __restrict__ idx,
                             int N, int total /*B*S*/, int S,
                             float* __restrict__ out)
{
    int gid = blockIdx.x * blockDim.x + threadIdx.x;
    if (gid >= total) return;
    int b = gid / S;
    int j = idx[gid];
    const float* p = xyz + ((size_t)b * N + j) * 3;
    out[(size_t)gid * 3 + 0] = p[0];
    out[(size_t)gid * 3 + 1] = p[1];
    out[(size_t)gid * 3 + 2] = p[2];
}

// ----------------------------- ball query ----------------------------------
// Reference semantics: ascending-index scan, first nsample points with
// d^2 <= r^2, padded with the first hit.
__global__ void k_ball(const float* __restrict__ xyz,
                       const float* __restrict__ newxyz,
                       int total /*B*S*/, int N, int S, int ns, float r2,
                       int* __restrict__ out)
{
    int gid = blockIdx.x * blockDim.x + threadIdx.x;
    if (gid >= total) return;
    int b = gid / S;
    const float* p = xyz + (size_t)b * N * 3;
    const float* q = newxyz + (size_t)gid * 3;
    const float qx = q[0], qy = q[1], qz = q[2];
    int* o = out + (size_t)gid * ns;
    int cnt = 0, first = 0;
    for (int j = 0; j < N && cnt < ns; ++j) {
        float dx = p[j * 3 + 0] - qx;
        float dy = p[j * 3 + 1] - qy;
        float dz = p[j * 3 + 2] - qz;
        if (dx * dx + dy * dy + dz * dz <= r2) {
            if (cnt == 0) first = j;
            o[cnt++] = j;
        }
    }
    for (; cnt < ns; ++cnt) o[cnt] = first;
}

// --------------------- grouped feature build (f16 X) -----------------------
// row = (b*S+s)*ns+k : [rel_xyz(3) | feat(Cf) | zero-pad to Kp]
__global__ void k_group(const float* __restrict__ xyz,
                        const float* __restrict__ newxyz,
                        const float* __restrict__ feat,
                        const int* __restrict__ idx,
                        int total /*B*S*ns*/, int N, int S, int ns, int Cf, int Kp,
                        h16* __restrict__ X)
{
    int gid = blockIdx.x * blockDim.x + threadIdx.x;
    if (gid >= total) return;
    int bs = gid / ns;                // b*S + s
    int b  = bs / S;
    int j  = idx[gid];
    const float* p = xyz + ((size_t)b * N + j) * 3;
    const float* q = newxyz + (size_t)bs * 3;
    const float* f = feat + ((size_t)b * N + j) * Cf;
    h16* xr = X + (size_t)gid * Kp;
    xr[0] = (h16)(p[0] - q[0]);
    xr[1] = (h16)(p[1] - q[1]);
    xr[2] = (h16)(p[2] - q[2]);
    int c = 3;
    for (int t = 0; t < Cf; ++t) xr[c++] = (h16)f[t];
    for (; c < Kp; ++c) xr[c] = (h16)0.0f;
}

// ----------------------- weight f32 -> f16 (K zero-pad) --------------------
__global__ void k_cvt_w(const float* __restrict__ w, int cout, int k, int kp,
                        h16* __restrict__ out)
{
    int gid = blockIdx.x * blockDim.x + threadIdx.x;
    if (gid >= cout * kp) return;
    int r = gid / kp, c = gid % kp;
    out[gid] = (c < k) ? (h16)w[(size_t)r * k + c] : (h16)0.0f;
}

// ------------------------------ WMMA GEMM ----------------------------------
// Y[M,Cout] = epilogue( X[M,Kp](f16) @ W[Cout,Kp](f16)^T ).
// Wave tile: 32(M) x 64(N) = 2 m-tiles x 4 n-tiles, 8 v8f accumulators.
// Per K=32 step: 2 A fragments + 4 B fragments -> 8 WMMAs (A reused 4x,
// B reused 2x). 8 waves / 256-thread block.
// A layout (16-bit 16x32): lanes 0-15 row M=lane hold K {0..7,16..23},
// lanes 16-31 hold K {8..15,24..31}. B column n = row n of W (same K order as A).
// C/D: lanes 0-15 -> N=lane, VGPR i -> M=i; lanes 16-31 -> N=lane-16, M=8+i.
__device__ __forceinline__ v16h ldA(const h16* __restrict__ p)
{
    v8h lo = *(const v8h*)p;
    v8h hi = *(const v8h*)(p + 16);
    v16h r;
#pragma unroll
    for (int i = 0; i < 8; ++i) { r[i] = lo[i]; r[i + 8] = hi[i]; }
    return r;
}
__device__ __forceinline__ v16h ldB(const h16* __restrict__ p)
{
    v8h lo = *(const v8h*)p;
    v8h hi = *(const v8h*)(p + 16);
    v16h r;
#pragma unroll
    for (int i = 0; i < 8; ++i) { r[i] = lo[i]; r[i + 8] = hi[i]; }
    return r;
}

__global__ void __launch_bounds__(256)
k_gemm(const h16* __restrict__ X, const h16* __restrict__ W,
       int M, int Kp, int Cout,
       const float* __restrict__ bias, const float* __restrict__ gamma,
       const float* __restrict__ beta, const float* __restrict__ mean,
       const float* __restrict__ var, int bn_relu,
       h16* __restrict__ outH, int ldh,
       float* __restrict__ outF, int ldf)
{
    const int n64   = Cout >> 6;               // 64-col groups
    const int tiles = (M >> 5) * n64;          // 32-row strips
    const int wave  = blockIdx.x * (blockDim.x >> 5) + (threadIdx.x >> 5);
    if (wave >= tiles) return;                 // wave-uniform exit (EXEC all-1)

    const int mt = wave / n64;
    const int nt = wave % n64;
    const int lane     = threadIdx.x & 31;
    const int half_sel = lane >> 4;
    const int lane16   = lane & 15;

    const h16* arow0 = X + (size_t)(mt * 32 + lane16) * Kp + half_sel * 8;
    const h16* arow1 = arow0 + (size_t)16 * Kp;
    const h16* brow  = W + (size_t)(nt * 64 + lane16) * Kp + half_sel * 8;
    const size_t bstep = (size_t)16 * Kp;

    v8f acc[2][4];
#pragma unroll
    for (int mi = 0; mi < 2; ++mi)
#pragma unroll
        for (int nj = 0; nj < 4; ++nj) acc[mi][nj] = (v8f)0.0f;

    for (int k0 = 0; k0 < Kp; k0 += 32) {
        v16h a0 = ldA(arow0 + k0);
        v16h a1 = ldA(arow1 + k0);
        v16h b0 = ldB(brow + k0);
        v16h b1 = ldB(brow + bstep + k0);
        v16h b2 = ldB(brow + 2 * bstep + k0);
        v16h b3 = ldB(brow + 3 * bstep + k0);
        acc[0][0] = __builtin_amdgcn_wmma_f32_16x16x32_f16(false, a0, false, b0, (short)0, acc[0][0], false, false);
        acc[1][0] = __builtin_amdgcn_wmma_f32_16x16x32_f16(false, a1, false, b0, (short)0, acc[1][0], false, false);
        acc[0][1] = __builtin_amdgcn_wmma_f32_16x16x32_f16(false, a0, false, b1, (short)0, acc[0][1], false, false);
        acc[1][1] = __builtin_amdgcn_wmma_f32_16x16x32_f16(false, a1, false, b1, (short)0, acc[1][1], false, false);
        acc[0][2] = __builtin_amdgcn_wmma_f32_16x16x32_f16(false, a0, false, b2, (short)0, acc[0][2], false, false);
        acc[1][2] = __builtin_amdgcn_wmma_f32_16x16x32_f16(false, a1, false, b2, (short)0, acc[1][2], false, false);
        acc[0][3] = __builtin_amdgcn_wmma_f32_16x16x32_f16(false, a0, false, b3, (short)0, acc[0][3], false, false);
        acc[1][3] = __builtin_amdgcn_wmma_f32_16x16x32_f16(false, a1, false, b3, (short)0, acc[1][3], false, false);
    }

    const int colbase = nt * 64 + lane16;
#pragma unroll
    for (int nj = 0; nj < 4; ++nj) {
        const int col = colbase + nj * 16;
        float s, t;
        if (bn_relu) {
            float g = gamma[col];
            s = g * rsqrtf(var[col] + 1e-5f);
            t = s * (bias[col] - mean[col]) + beta[col];
        } else {
            s = 1.0f; t = bias[col];
        }
#pragma unroll
        for (int mi = 0; mi < 2; ++mi) {
            const int mbase = mt * 32 + mi * 16 + half_sel * 8;
#pragma unroll
            for (int i = 0; i < 8; ++i) {
                float y = acc[mi][nj][i] * s + t;
                if (bn_relu) y = fmaxf(y, 0.0f);
                size_t row = (size_t)(mbase + i);
                if (outH) outH[row * (size_t)ldh + col] = (h16)y;
                if (outF) outF[row * (size_t)ldf + col] = y;
            }
        }
    }
}

// ------------------------------ max pool -----------------------------------
__global__ void k_maxpool(const h16* __restrict__ in, int G, int ns, int C,
                          float* __restrict__ out)
{
    int gid = blockIdx.x * blockDim.x + threadIdx.x;
    if (gid >= G * C) return;
    int g = gid / C, c = gid % C;
    const h16* p = in + (size_t)g * ns * C + c;
    float m = -1e30f;
    for (int k = 0; k < ns; ++k) m = fmaxf(m, (float)p[(size_t)k * C]);
    out[(size_t)g * C + c] = m;
}

// ----------------------- sa3 build: [xyz | pts | pad] ----------------------
__global__ void k_concat2(const float* __restrict__ A, int Ca,
                          const float* __restrict__ Bp, int Cb,
                          int R, int Kp, h16* __restrict__ X)
{
    int r = blockIdx.x * blockDim.x + threadIdx.x;
    if (r >= R) return;
    h16* xr = X + (size_t)r * Kp;
    const float* a = A + (size_t)r * Ca;
    const float* b = Bp + (size_t)r * Cb;
    int c = 0;
    for (int j = 0; j < Ca; ++j) xr[c++] = (h16)a[j];
    for (int j = 0; j < Cb; ++j) xr[c++] = (h16)b[j];
    for (; c < Kp; ++c) xr[c] = (h16)0.0f;
}

// ----------------- fp3 build: [l2pts(256) | bcast l3pts(1024)] -------------
__global__ void k_build_fp3(const float* __restrict__ l2pts,
                            const float* __restrict__ l3pts,
                            int R, h16* __restrict__ X)
{
    int r = blockIdx.x * blockDim.x + threadIdx.x;
    if (r >= R) return;
    int b = r / 128;
    h16* xr = X + (size_t)r * 1280;
    const float* a = l2pts + (size_t)r * 256;
    const float* g = l3pts + (size_t)b * 1024;
    for (int j = 0; j < 256; ++j)  xr[j]       = (h16)a[j];
    for (int j = 0; j < 1024; ++j) xr[256 + j] = (h16)g[j];
}

// ------------------------------ 3-NN ---------------------------------------
__global__ void k_knn3(const float* __restrict__ xyz1,
                       const float* __restrict__ xyz2,
                       int total /*B*S1*/, int S1, int S2,
                       int* __restrict__ idx, float* __restrict__ w)
{
    int gid = blockIdx.x * blockDim.x + threadIdx.x;
    if (gid >= total) return;
    int b = gid / S1;
    const float* q = xyz1 + (size_t)gid * 3;
    const float* p = xyz2 + (size_t)b * S2 * 3;
    const float qx = q[0], qy = q[1], qz = q[2];
    float d0 = 3e38f, d1 = 3e38f, d2 = 3e38f;
    int   i0 = 0,     i1 = 0,     i2 = 0;
    for (int j = 0; j < S2; ++j) {
        float dx = p[j * 3 + 0] - qx;
        float dy = p[j * 3 + 1] - qy;
        float dz = p[j * 3 + 2] - qz;
        float d  = dx * dx + dy * dy + dz * dz;
        if (d < d0)      { d2 = d1; i2 = i1; d1 = d0; i1 = i0; d0 = d; i0 = j; }
        else if (d < d1) { d2 = d1; i2 = i1; d1 = d;  i1 = j; }
        else if (d < d2) { d2 = d;  i2 = j; }
    }
    float w0 = 1.0f / (d0 + 1e-8f);
    float w1 = 1.0f / (d1 + 1e-8f);
    float w2 = 1.0f / (d2 + 1e-8f);
    float sum = w0 + w1 + w2;
    idx[(size_t)gid * 3 + 0] = i0;
    idx[(size_t)gid * 3 + 1] = i1;
    idx[(size_t)gid * 3 + 2] = i2;
    w[(size_t)gid * 3 + 0] = w0 / sum;
    w[(size_t)gid * 3 + 1] = w1 / sum;
    w[(size_t)gid * 3 + 2] = w2 / sum;
}

// ----------- fp build: [p1a | p1b? | 3-NN weighted interp | pad] -----------
__global__ void k_build_fp(const float* __restrict__ p1a, int Ca,
                           const float* __restrict__ p1b, int Cb,
                           const h16* __restrict__ p2, int S2, int C2,
                           const int* __restrict__ idx,
                           const float* __restrict__ w,
                           int R, int S1, int Kp, h16* __restrict__ X)
{
    int r = blockIdx.x * blockDim.x + threadIdx.x;
    if (r >= R) return;
    int b = r / S1;
    h16* xr = X + (size_t)r * Kp;
    int c = 0;
    const float* a = p1a + (size_t)r * Ca;
    for (int j = 0; j < Ca; ++j) xr[c++] = (h16)a[j];
    if (p1b) {
        const float* bb = p1b + (size_t)r * Cb;
        for (int j = 0; j < Cb; ++j) xr[c++] = (h16)bb[j];
    }
    int j0 = idx[(size_t)r * 3 + 0];
    int j1 = idx[(size_t)r * 3 + 1];
    int j2 = idx[(size_t)r * 3 + 2];
    float w0 = w[(size_t)r * 3 + 0];
    float w1 = w[(size_t)r * 3 + 1];
    float w2 = w[(size_t)r * 3 + 2];
    const h16* q0 = p2 + ((size_t)b * S2 + j0) * C2;
    const h16* q1 = p2 + ((size_t)b * S2 + j1) * C2;
    const h16* q2 = p2 + ((size_t)b * S2 + j2) * C2;
    for (int k = 0; k < C2; ++k)
        xr[c++] = (h16)(w0 * (float)q0[k] + w1 * (float)q1[k] + w2 * (float)q2[k]);
    for (; c < Kp; ++c) xr[c] = (h16)0.0f;
}

// ===========================================================================
// Host side
// ===========================================================================
struct Lyr { const float *w, *b, *g, *be, *mu, *va; };

static inline int cdiv(int a, int b) { return (a + b - 1) / b; }

extern "C" void kernel_launch(void* const* d_in, const int* in_sizes, int n_in,
                              void* d_out, int out_size, void* d_ws, size_t ws_size,
                              hipStream_t stream)
{
    (void)in_sizes; (void)n_in; (void)out_size;

    const int B = 8, N = 8192;

    // ---- input pointers (setup_inputs dict insertion order) ----
    const float* xyz = (const float*)d_in[0];
    auto getL = [&](int base) -> Lyr {
        return Lyr{ (const float*)d_in[base + 0], (const float*)d_in[base + 1],
                    (const float*)d_in[base + 2], (const float*)d_in[base + 3],
                    (const float*)d_in[base + 4], (const float*)d_in[base + 5] };
    };
    Lyr sa1[3]  = { getL(1),  getL(7),  getL(13) };
    Lyr sa2[3]  = { getL(19), getL(25), getL(31) };
    Lyr sa3[3]  = { getL(37), getL(43), getL(49) };
    Lyr fp3[2]  = { getL(55), getL(61) };
    Lyr fp2[2]  = { getL(67), getL(73) };
    Lyr fp1[3]  = { getL(79), getL(85), getL(91) };
    Lyr head1   = getL(97);
    const float* h2w = (const float*)d_in[103];
    const float* h2b = (const float*)d_in[104];

    // ---- workspace bump allocator ----
    char* ws = (char*)d_ws;
    size_t off = 0;
    auto alloc = [&](size_t bytes) -> char* {
        char* p = ws + off;
        off = (off + bytes + 255) & ~(size_t)255;
        return p;
    };

    h16*   PA      = (h16*)alloc((size_t)131072 * 128 * sizeof(h16));
    h16*   PB      = (h16*)alloc((size_t)131072 * 128 * sizeof(h16));
    float* l1xyz   = (float*)alloc((size_t)B * 512 * 3 * sizeof(float));
    float* l2xyz   = (float*)alloc((size_t)B * 128 * 3 * sizeof(float));
    float* l1pts   = (float*)alloc((size_t)B * 512 * 128 * sizeof(float));
    float* l2pts   = (float*)alloc((size_t)B * 128 * 256 * sizeof(float));
    float* l3pts   = (float*)alloc((size_t)B * 1024 * sizeof(float));
    h16*   l2pnewh = (h16*)alloc((size_t)B * 128 * 256 * sizeof(h16));
    h16*   l1pnewh = (h16*)alloc((size_t)B * 512 * 128 * sizeof(h16));
    int*   fps1    = (int*)alloc((size_t)B * 512 * sizeof(int));
    int*   fps2    = (int*)alloc((size_t)B * 128 * sizeof(int));
    int*   ball1   = (int*)alloc((size_t)B * 512 * 32 * sizeof(int));
    int*   ball2   = (int*)alloc((size_t)B * 128 * 64 * sizeof(int));
    int*   knn2i   = (int*)alloc((size_t)B * 512 * 3 * sizeof(int));
    float* knn2w   = (float*)alloc((size_t)B * 512 * 3 * sizeof(float));
    int*   knn1i   = (int*)alloc((size_t)B * 8192 * 3 * sizeof(int));
    float* knn1w   = (float*)alloc((size_t)B * 8192 * 3 * sizeof(float));

    // f16 weights, K zero-padded to multiple of 32
    struct WS { const float* src; int cout, cin, kp; h16* dst; };
    WS w[17] = {
        { sa1[0].w,   64,    6,   32, nullptr }, { sa1[1].w,   64,   64,   64, nullptr },
        { sa1[2].w,  128,   64,   64, nullptr }, { sa2[0].w,  128,  131,  160, nullptr },
        { sa2[1].w,  128,  128,  128, nullptr }, { sa2[2].w,  256,  128,  128, nullptr },
        { sa3[0].w,  256,  259,  288, nullptr }, { sa3[1].w,  512,  256,  256, nullptr },
        { sa3[2].w, 1024,  512,  512, nullptr }, { fp3[0].w,  256, 1280, 1280, nullptr },
        { fp3[1].w,  256,  256,  256, nullptr }, { fp2[0].w,  256,  384,  384, nullptr },
        { fp2[1].w,  128,  256,  256, nullptr }, { fp1[0].w,  128,  134,  160, nullptr },
        { fp1[1].w,  128,  128,  128, nullptr }, { fp1[2].w,  128,  128,  128, nullptr },
        { head1.w,   128,  128,  128, nullptr }
    };
    for (int i = 0; i < 17; ++i)
        w[i].dst = (h16*)alloc((size_t)w[i].cout * w[i].kp * sizeof(h16));
    h16* h2wh = (h16*)alloc((size_t)128 * 128 * sizeof(h16));
    if (off > ws_size) return;   // workspace too small: do nothing

    // ---- convert weights ----
    for (int i = 0; i < 17; ++i) {
        int n = w[i].cout * w[i].kp;
        k_cvt_w<<<cdiv(n, 256), 256, 0, stream>>>(w[i].src, w[i].cout, w[i].cin,
                                                  w[i].kp, w[i].dst);
    }
    k_cvt_w<<<cdiv(128 * 128, 256), 256, 0, stream>>>(h2w, 128, 128, 128, h2wh);

    auto gemm = [&](const h16* X, h16* Wh, int M, int Kp, int Cout,
                    const Lyr& L, int bn, h16* oH, float* oF, int ldf) {
        int tiles = (M / 32) * (Cout / 64);    // 32x64 wave tiles
        k_gemm<<<cdiv(tiles, 8), 256, 0, stream>>>(
            X, Wh, M, Kp, Cout, L.b, L.g, L.be, L.mu, L.va, bn,
            oH, Cout, oF, ldf);
    };

    // ================= SA1: N=8192 -> 512 centers, ns=32 ===================
    k_fps<<<B, 256, 0, stream>>>(xyz, N, 512, fps1);
    k_gather_xyz<<<cdiv(B * 512, 256), 256, 0, stream>>>(xyz, fps1, N, B * 512, 512, l1xyz);
    k_ball<<<cdiv(B * 512, 256), 256, 0, stream>>>(xyz, l1xyz, B * 512, N, 512, 32,
                                                   0.2f * 0.2f, ball1);
    k_group<<<cdiv(131072, 256), 256, 0, stream>>>(xyz, l1xyz, xyz, ball1,
                                                   131072, N, 512, 32, 3, 32, PA);
    gemm(PA, w[0].dst, 131072, 32,  64, sa1[0], 1, PB, nullptr, 0);
    gemm(PB, w[1].dst, 131072, 64,  64, sa1[1], 1, PA, nullptr, 0);
    gemm(PA, w[2].dst, 131072, 64, 128, sa1[2], 1, PB, nullptr, 0);
    k_maxpool<<<cdiv(4096 * 128, 256), 256, 0, stream>>>(PB, 4096, 32, 128, l1pts);

    // ================= SA2: 512 -> 128 centers, ns=64 ======================
    k_fps<<<B, 256, 0, stream>>>(l1xyz, 512, 128, fps2);
    k_gather_xyz<<<cdiv(B * 128, 256), 256, 0, stream>>>(l1xyz, fps2, 512, B * 128, 128, l2xyz);
    k_ball<<<cdiv(B * 128, 256), 256, 0, stream>>>(l1xyz, l2xyz, B * 128, 512, 128, 64,
                                                   0.4f * 0.4f, ball2);
    k_group<<<cdiv(65536, 256), 256, 0, stream>>>(l1xyz, l2xyz, l1pts, ball2,
                                                  65536, 512, 128, 64, 128, 160, PA);
    gemm(PA, w[3].dst, 65536, 160, 128, sa2[0], 1, PB, nullptr, 0);
    gemm(PB, w[4].dst, 65536, 128, 128, sa2[1], 1, PA, nullptr, 0);
    gemm(PA, w[5].dst, 65536, 128, 256, sa2[2], 1, PB, nullptr, 0);
    k_maxpool<<<cdiv(1024 * 256, 256), 256, 0, stream>>>(PB, 1024, 64, 256, l2pts);

    // ================= SA3: group-all over 128 points ======================
    k_concat2<<<cdiv(1024, 256), 256, 0, stream>>>(l2xyz, 3, l2pts, 256, 1024, 288, PA);
    gemm(PA, w[6].dst, 1024, 288,  256, sa3[0], 1, PB, nullptr, 0);
    gemm(PB, w[7].dst, 1024, 256,  512, sa3[1], 1, PA, nullptr, 0);
    gemm(PA, w[8].dst, 1024, 512, 1024, sa3[2], 1, PB, nullptr, 0);
    k_maxpool<<<cdiv(8 * 1024, 256), 256, 0, stream>>>(PB, 8, 128, 1024, l3pts);

    // ================= FP3: broadcast l3 -> l2 =============================
    k_build_fp3<<<cdiv(1024, 256), 256, 0, stream>>>(l2pts, l3pts, 1024, PA);
    gemm(PA, w[9].dst,  1024, 1280, 256, fp3[0], 1, PB, nullptr, 0);
    gemm(PB, w[10].dst, 1024,  256, 256, fp3[1], 1, l2pnewh, nullptr, 0);

    // ================= FP2: 3-NN l2 -> l1 ==================================
    k_knn3<<<cdiv(B * 512, 256), 256, 0, stream>>>(l1xyz, l2xyz, B * 512, 512, 128,
                                                   knn2i, knn2w);
    k_build_fp<<<cdiv(4096, 256), 256, 0, stream>>>(l1pts, 128, nullptr, 0,
                                                    l2pnewh, 128, 256, knn2i, knn2w,
                                                    4096, 512, 384, PA);
    gemm(PA, w[11].dst, 4096, 384, 256, fp2[0], 1, PB, nullptr, 0);
    gemm(PB, w[12].dst, 4096, 256, 128, fp2[1], 1, l1pnewh, nullptr, 0);

    // ================= FP1: 3-NN l1 -> l0 ==================================
    k_knn3<<<cdiv(B * 8192, 256), 256, 0, stream>>>(xyz, l1xyz, B * 8192, 8192, 512,
                                                    knn1i, knn1w);
    // points1 = concat(l0_xyz, l0_points=xyz) -> 6 channels
    k_build_fp<<<cdiv(65536, 256), 256, 0, stream>>>(xyz, 3, xyz, 3,
                                                     l1pnewh, 512, 128, knn1i, knn1w,
                                                     65536, 8192, 160, PA);
    gemm(PA, w[13].dst, 65536, 160, 128, fp1[0], 1, PB, nullptr, 0);
    gemm(PB, w[14].dst, 65536, 128, 128, fp1[1], 1, PA, nullptr, 0);
    gemm(PA, w[15].dst, 65536, 128, 128, fp1[2], 1, PB, nullptr, 0);

    // ================= head ===============================================
    gemm(PB, w[16].dst, 65536, 128, 128, head1, 1, PA, nullptr, 0);
    // head2: plain linear, f32 straight into d_out (8,8192,128)
    Lyr h2 = { h2w, h2b, nullptr, nullptr, nullptr, nullptr };
    gemm(PA, h2wh, 65536, 128, 128, h2, 0, nullptr, (float*)d_out, 128);
}
